// CriticTanh_90263032693203
// MI455X (gfx1250) — matrix operand order repaired and checked by reading, not compiled, not run on hardware
//
#include <hip/hip_runtime.h>
#include <math.h>

typedef __attribute__((ext_vector_type(16))) _Float16 v16h;
typedef __attribute__((ext_vector_type(8)))  float    v8f;

#define BS  1024
#define NN  21
#define KK  20
#define HID 128
#define EMB 64
#define DD  192
#define NNODE (BS * NN)

// CDNA5 has a hardware transcendental V_TANH_F32 (TRANS32 pipe, co-executes
// with WMMA).  Use it if the toolchain declares the builtin; otherwise fall
// back to libm's polynomial expansion.
__device__ __forceinline__ float ftanh(float x) {
#if __has_builtin(__builtin_amdgcn_tanhf)
    return __builtin_amdgcn_tanhf(x);
#else
    return tanhf(x);
#endif
}

struct HeadParams {
    const float* sW1; const float* sb1; const float* sW2; const float* sb2;
    const float* emb; const float* eW;  const float* eb;
    const float* mW1; const float* mb1; const float* mW2; const float* mb2;
    const float* tW1; const float* tb1; const float* tW2; const float* tb2;
};

// ---------------------------------------------------------------------------
// Kernel 1: per-node feature pipeline.  h = tanh([se | ce]) (192), then the
// factored message layer-1:  u = h@(A-B)+mb1,  v = h@B   (A,B = halves of mW1)
// One block (128 threads) per node.
// ---------------------------------------------------------------------------
__global__ __launch_bounds__(HID)
void node_kernel(const float* __restrict__ positions,
                 const int*   __restrict__ categories,
                 const float* __restrict__ action,
                 const float* __restrict__ tar_scores,
                 HeadParams P,
                 float* __restrict__ u, float* __restrict__ v)
{
    const int node = blockIdx.x;
    const int t    = threadIdx.x;
    const int b    = node / NN;
    const int ni   = node % NN;

    __shared__ float se1[HID];
    __shared__ float hbuf[DD];

    // spatial input (6): [pos.x, pos.y, act0, act1, tanh(ts0), tanh(ts1)]
    const float sp0 = positions[2 * node + 0];
    const float sp1 = positions[2 * node + 1];
    const float sp2 = action[b * (NN * 2) + 2 * ni + 0];
    const float sp3 = action[b * (NN * 2) + 2 * ni + 1];
    const float sp4 = ftanh(tar_scores[2 * node + 0]);
    const float sp5 = ftanh(tar_scores[2 * node + 1]);

    // layer 1 of spatial MLP (6 -> 128)
    float acc = P.sb1[t];
    acc += sp0 * P.sW1[0 * HID + t];
    acc += sp1 * P.sW1[1 * HID + t];
    acc += sp2 * P.sW1[2 * HID + t];
    acc += sp3 * P.sW1[3 * HID + t];
    acc += sp4 * P.sW1[4 * HID + t];
    acc += sp5 * P.sW1[5 * HID + t];
    se1[t] = ftanh(acc);
    __syncthreads();

    // layer 2 of spatial MLP (128 -> 128)
    float se = P.sb2[t];
#pragma unroll 4
    for (int k = 0; k < HID; ++k) se += se1[k] * P.sW2[k * HID + t];
    hbuf[t] = ftanh(se);

    // category embedding branch (64 -> 64), threads 0..63
    if (t < EMB) {
        const int cat = categories[node];
        float ce = P.eb[t];
#pragma unroll 4
        for (int k = 0; k < EMB; ++k)
            ce += ftanh(P.emb[cat * EMB + k]) * P.eW[k * EMB + t];
        hbuf[HID + t] = ftanh(ce);
    }
    __syncthreads();

    // factored message layer 1: u = h@(A-B) + mb1 ; v = h@B
    float uacc = P.mb1[t];
    float vacc = 0.0f;
#pragma unroll 4
    for (int k = 0; k < DD; ++k) {
        const float hk = hbuf[k];
        const float wa = P.mW1[k * HID + t];          // rows 0..191  (xi)
        const float wb = P.mW1[(DD + k) * HID + t];   // rows 192..383 (xj-xi)
        uacc += hk * (wa - wb);
        vacc += hk * wb;
    }
    u[node * HID + t] = uacc;
    v[node * HID + t] = vacc;
}

// ---------------------------------------------------------------------------
// Kernel 2: per-pair layer-2 GEMM + max-over-neighbors, via WMMA f16.
// One block per batch (256 threads = 8 wave32).  Wave w owns N-columns
// [16w, 16w+16).  For each node i: A-tile rows = tanh(u_i + v_j) for the 20
// neighbors j != i (f16, rows 20..31 zero-padded); B = mW2 tile held in
// VGPRs across the whole i-loop; 4 k-steps x 2 M-tiles of
// v_wmma_f32_16x16x32_f16; then elementwise max over valid rows, bias, tanh.
// ---------------------------------------------------------------------------
#define A_STRIDE 136   // halves per row; 272 bytes keeps 16B alignment + skews banks

__global__ __launch_bounds__(256)
void pair_kernel(const float* __restrict__ u, const float* __restrict__ v,
                 const float* __restrict__ mW2, const float* __restrict__ mb2,
                 float* __restrict__ agg)
{
    const int b    = blockIdx.x;
    const int tid  = threadIdx.x;
    const int wave = tid >> 5;
    const int lane = tid & 31;
    const int half16 = lane >> 4;          // 0: lanes 0-15, 1: lanes 16-31
    const int ncol   = (wave << 4) + (lane & 15);

    __shared__ float    uS[NN][HID];
    __shared__ float    vS[NN][HID];
    __shared__ _Float16 tA[32][A_STRIDE];

    // stage u, v for this batch into LDS
    for (int idx = tid; idx < NN * HID; idx += 256) {
        const int n0 = idx >> 7, c = idx & (HID - 1);
        uS[n0][c] = u[(b * NN + n0) * HID + c];
        vS[n0][c] = v[(b * NN + n0) * HID + c];
    }

    // preload B fragments (mW2 is invariant across i and batches)
    // B k order used here (contiguous, as in A): lane -> column, element e -> K = 32*ks + 16*half + e
    v16h bf[4];
#pragma unroll
    for (int ks = 0; ks < 4; ++ks) {
        const int kb = ks * 32 + half16 * 16;
#pragma unroll
        for (int e = 0; e < 16; ++e)
            bf[ks][e] = (_Float16)mW2[(kb + e) * HID + ncol];
    }
    __syncthreads();

    for (int i = 0; i < NN; ++i) {
        // build A tile: rows 0..19 = tanh(u_i + v_j), j runs over all j != i
        for (int idx = tid; idx < 32 * HID; idx += 256) {
            const int r = idx >> 7, c = idx & (HID - 1);
            float val = 0.0f;
            if (r < KK) {
                const int j = (r < i) ? r : (r + 1);
                val = ftanh(uS[i][c] + vS[j][c]);
            }
            tA[r][c] = (_Float16)val;
        }
        __syncthreads();

        v8f c0 = {};   // rows 0..15
        v8f c1 = {};   // rows 16..31 (rows 16..19 valid)
        const int arow = lane & 15;
#pragma unroll
        for (int ks = 0; ks < 4; ++ks) {
            // A k order used here (contiguous, as in B): lane half picks K {0..15} / {16..31}
            const int kb = ks * 32 + (half16 ? 16 : 0);
            v16h a0, a1;
#pragma unroll
            for (int e = 0; e < 8; ++e) {
                a0[e]     = tA[arow][kb + e];
                a0[e + 8] = tA[arow][kb + 8 + e];
                a1[e]     = tA[arow + 16][kb + e];
                a1[e + 8] = tA[arow + 16][kb + 8 + e];
            }
            c0 = __builtin_amdgcn_wmma_f32_16x16x32_f16(
                     false, a0, false, bf[ks], (short)0, c0, false, false);
            c1 = __builtin_amdgcn_wmma_f32_16x16x32_f16(
                     false, a1, false, bf[ks], (short)0, c1, false, false);
        }

        // D layout: VGPR r, lanes 0-15 -> row r ; lanes 16-31 -> row 8+r.
        // lane<16 owns rows 0..7 (c0) and 16..23 (c1, rows 16..19 valid);
        // lane>=16 owns rows 8..15 (c0).  Max-reduce over valid rows.
        float m = -3.0e38f;
#pragma unroll
        for (int r = 0; r < 8; ++r) m = fmaxf(m, c0[r]);
        if (!half16) {
#pragma unroll
            for (int r = 0; r < 4; ++r) m = fmaxf(m, c1[r]);
        }
        m = fmaxf(m, __shfl_xor(m, 16, 32));   // merge the lane-half pair

        if (!half16)
            agg[(b * NN + i) * HID + ncol] = ftanh(m + mb2[ncol]);
        __syncthreads();
    }
}

// ---------------------------------------------------------------------------
// Kernel 3: final head MLP  tanh(agg@tW1 + tb1) @ tW2 + tb2 -> scalar/node.
// ---------------------------------------------------------------------------
__global__ __launch_bounds__(HID)
void final_kernel(const float* __restrict__ agg, HeadParams P,
                  float* __restrict__ out)
{
    const int node = blockIdx.x;
    const int t    = threadIdx.x;
    __shared__ float arow[HID];
    __shared__ float red[HID];

    arow[t] = agg[node * HID + t];
    __syncthreads();

    float acc = P.tb1[t];
#pragma unroll 4
    for (int k = 0; k < HID; ++k) acc += arow[k] * P.tW1[k * HID + t];
    red[t] = ftanh(acc) * P.tW2[t];
    __syncthreads();

    for (int s = HID / 2; s > 0; s >>= 1) {
        if (t < s) red[t] += red[t + s];
        __syncthreads();
    }
    if (t == 0) out[node] = red[0] + P.tb2[0];
}

// ---------------------------------------------------------------------------
// Launch
// ---------------------------------------------------------------------------
static HeadParams make_head(void* const* leaves)
{
    // jax tree order: keys sorted alphabetically within each head dict:
    // eW, eb, emb, mW1, mW2, mb1, mb2, sW1, sW2, sb1, sb2, tW1, tW2, tb1, tb2
    HeadParams P;
    P.eW  = (const float*)leaves[0];
    P.eb  = (const float*)leaves[1];
    P.emb = (const float*)leaves[2];
    P.mW1 = (const float*)leaves[3];
    P.mW2 = (const float*)leaves[4];
    P.mb1 = (const float*)leaves[5];
    P.mb2 = (const float*)leaves[6];
    P.sW1 = (const float*)leaves[7];
    P.sW2 = (const float*)leaves[8];
    P.sb1 = (const float*)leaves[9];
    P.sb2 = (const float*)leaves[10];
    P.tW1 = (const float*)leaves[11];
    P.tW2 = (const float*)leaves[12];
    P.tb1 = (const float*)leaves[13];
    P.tb2 = (const float*)leaves[14];
    return P;
}

extern "C" void kernel_launch(void* const* d_in, const int* in_sizes, int n_in,
                              void* d_out, int out_size, void* d_ws, size_t ws_size,
                              hipStream_t stream)
{
    const float* positions;
    const int*   categories;
    const float* action;
    const float* tar_scores;
    int pbase;

    // Two plausible flattenings of setup_inputs():
    //  insertion order: positions, categories, action, tar_scores, params(30)
    //  sorted tree:     action, categories, params(30), positions, tar_scores
    // in_sizes[2] is 4096 (h1.eW) only in the sorted layout (else 43008).
    if (n_in >= 34 && in_sizes[2] == 4096) {
        action     = (const float*)d_in[0];
        categories = (const int*)  d_in[1];
        pbase      = 2;
        positions  = (const float*)d_in[32];
        tar_scores = (const float*)d_in[33];
    } else {
        positions  = (const float*)d_in[0];
        categories = (const int*)  d_in[1];
        action     = (const float*)d_in[2];
        tar_scores = (const float*)d_in[3];
        pbase      = 4;
    }

    // workspace: U (u, later aliased by agg) and V — 2 x 21504 x 128 f32
    float* U = (float*)d_ws;
    float* V = U + (size_t)NNODE * HID;
    float* AGG = U;   // safe alias: pair_kernel stages u into LDS before writing agg

    float* out = (float*)d_out;

    for (int h = 0; h < 2; ++h) {
        HeadParams P = make_head(d_in + pbase + h * 15);

        node_kernel<<<NNODE, HID, 0, stream>>>(positions, categories, action,
                                               tar_scores, P, U, V);
        pair_kernel<<<BS, 256, 0, stream>>>(U, V, P.mW2, P.mb2, AGG);
        final_kernel<<<NNODE, HID, 0, stream>>>(AGG, P, out + (size_t)h * NNODE);
    }
}
